// GraphAttentionLayer_86912958202744
// MI455X (gfx1250) — matrix operand order repaired and checked by reading, not compiled, not run on hardware
//
#include <hip/hip_runtime.h>

typedef __attribute__((ext_vector_type(16))) __bf16 v16bf;
typedef __attribute__((ext_vector_type(8)))  float  v8f;

#define ALPHA_   0.2f
#define NEG_BIG  -9.0e15f

#define B_     8
#define CIN    32
#define NN     2048
#define LL     12
#define FO     32
#define NTILES 24      /* (LL*FO)/16 */
#define KTILES 64      /* NN/32     */
#define WTILES 128     /* NN/16     */

__device__ __forceinline__ unsigned short f32_to_bf16(float x) {
  unsigned u = __float_as_uint(x);
  u += 0x7FFFu + ((u >> 16) & 1u);          // round-to-nearest-even
  return (unsigned short)(u >> 16);
}

// gfx1250 async Global<->LDS DMA (ASYNCcnt) ---------------------------------
__device__ __forceinline__ void async_load_b128(unsigned lds_off, const void* g) {
  asm volatile("global_load_async_to_lds_b128 %0, %1, off"
               :: "v"(lds_off), "v"(g) : "memory");
}
__device__ __forceinline__ void async_store_b128(void* g, unsigned lds_off) {
  asm volatile("global_store_async_from_lds_b128 %0, %1, off"
               :: "v"(g), "v"(lds_off) : "memory");
}
__device__ __forceinline__ void wait_async0() {
  asm volatile("s_wait_asynccnt 0" ::: "memory");
}
__device__ __forceinline__ unsigned lds_offset(const void* p) {
  return (unsigned)(unsigned long long)p;   // generic->LDS: low 32 bits
}

// ---------------------------------------------------------------------------
// Kernel 1: Wh[b,l,v,f] = sum_c h[b,c,v,l] * W[c,f]
// Writes Wh as bf16 in WMMA *B-fragment* order: frag(b,kk,ntile)[lane*16+j],
//   lane = half*16 + (f%16), half=(v%32)/16, j=v%16, ntile = l*2 + f/16.
// Also emits s1[b,v], s2[b,v] from the l==11 slice (Wh_last @ a).
// ---------------------------------------------------------------------------
__global__ __launch_bounds__(384)
void k1_wh(const float* __restrict__ h, const float* __restrict__ W,
           const float* __restrict__ a, unsigned short* __restrict__ Bfrag,
           float* __restrict__ s1, float* __restrict__ s2) {
  __shared__ float sW[CIN * FO];
  __shared__ float sA[2 * FO];
  __shared__ unsigned short sTile[NTILES * 512];   // 24 KB, frag-ordered

  const int t  = threadIdx.x;
  const int vt = blockIdx.x;        // k-tile of 32 nodes
  const int b  = blockIdx.y;

  for (int i = t; i < CIN * FO; i += 384) sW[i] = W[i];
  if (t < 2 * FO) sA[t] = a[t];
  __syncthreads();

  const int vloc = t / LL;          // 0..31
  const int l    = t % LL;          // 0..11
  const int v    = vt * 32 + vloc;

  float acc[FO];
  #pragma unroll
  for (int f = 0; f < FO; ++f) acc[f] = 0.f;

  // h index = ((b*32+c)*2048+v)*12+l ; over threads, v*12+l == t -> coalesced
  const float* hp = h + (((size_t)b * CIN) * NN + v) * LL + l;
  for (int c = 0; c < CIN; ++c) {
    float hv = hp[(size_t)c * NN * LL];
    #pragma unroll
    for (int f = 0; f < FO; ++f) acc[f] = fmaf(hv, sW[c * FO + f], acc[f]);
  }

  const int half = vloc >> 4, j = vloc & 15;
  #pragma unroll
  for (int f = 0; f < FO; ++f) {
    int ntile = l * 2 + (f >> 4);
    int lane  = half * 16 + (f & 15);
    sTile[ntile * 512 + lane * 16 + j] = f32_to_bf16(acc[f]);
  }

  if (l == LL - 1) {                 // Wh_last row -> attention logits
    float d1 = 0.f, d2 = 0.f;
    #pragma unroll
    for (int f = 0; f < FO; ++f) {
      d1 = fmaf(acc[f], sA[f], d1);
      d2 = fmaf(acc[f], sA[FO + f], d2);
    }
    s1[b * NN + v] = d1;
    s2[b * NN + v] = d2;
  }
  __syncthreads();

  // coalesced 24KB dump via async LDS->global DMA
  uint4* dst = (uint4*)(Bfrag + ((size_t)(b * KTILES + vt) * NTILES) * 512);
  const unsigned sbase = lds_offset(sTile);
  for (int i = t; i < NTILES * 64; i += 384)
    async_store_b128(dst + i, sbase + i * 16);
  wait_async0();
}

// ---------------------------------------------------------------------------
// Kernel 2: column softmax of masked leaky-relu scores; writes attnT[w=j][v=i]
// directly in WMMA *A-fragment* order (contiguous k order: lanes 0-15 hold
// K in {0..15}, lanes 16-31 hold K in {16..31}; same k order as the B frags).
// One block = 64 j-columns of one batch; adj reads coalesced over j.
// ---------------------------------------------------------------------------
__global__ __launch_bounds__(256)
void k2_attn(const int* __restrict__ adj, const float* __restrict__ s1g,
             const float* __restrict__ s2g, unsigned short* __restrict__ Afrag) {
  __shared__ float s1sh[NN];                 // 8 KB
  __shared__ float redM[256], redS[256];
  __shared__ float colM[64], colR[64];
  __shared__ unsigned short sFrag[4 * 512];  // 4 frags (4 w-tiles) per kk

  const int t  = threadIdx.x;
  const int jb = blockIdx.x * 64;
  const int b  = blockIdx.y;

  for (int i = t; i < NN; i += 256) s1sh[i] = s1g[b * NN + i];
  __syncthreads();

  const int tj = t & 63, ti = t >> 6;        // tj fast -> coalesced adj
  const int jcol = jb + tj;
  const float s2j = s2g[b * NN + jcol];

  // pass 1: online max/sum over i (softmax axis)
  float mx = -3.0e38f, sm = 0.f;
  for (int v = ti; v < NN; v += 4) {
    int av   = adj[(size_t)v * NN + jcol];
    float sv = s1sh[v] + s2j;
    float e  = sv > 0.f ? sv : ALPHA_ * sv;
    float vl = av > 0 ? e : NEG_BIG;
    float nm = fmaxf(mx, vl);
    sm = sm * __expf(mx - nm) + __expf(vl - nm);
    mx = nm;
  }
  redM[t] = mx; redS[t] = sm;
  __syncthreads();
  if (t < 64) {
    float M = redM[t], S = redS[t];
    #pragma unroll
    for (int p = 1; p < 4; ++p) {
      float m2 = redM[t + 64 * p], sp = redS[t + 64 * p];
      float nm = fmaxf(M, m2);
      S = S * __expf(M - nm) + sp * __expf(m2 - nm);
      M = nm;
    }
    colM[t] = M; colR[t] = 1.0f / S;
  }
  __syncthreads();

  // pass 2: normalized weights -> A-fragment swizzle in LDS -> async DMA dump
  const float M = colM[tj], R = colR[tj];
  const int wloc = tj >> 4, m = tj & 15;
  const int half = ti >> 1, jel0 = (ti & 1) * 8;
  const unsigned fbase = lds_offset(sFrag);
  for (int kk = 0; kk < KTILES; ++kk) {
    #pragma unroll
    for (int s = 0; s < 8; ++s) {
      int v    = kk * 32 + ti * 8 + s;
      int av   = adj[(size_t)v * NN + jcol];
      float sv = s1sh[v] + s2j;
      float e  = sv > 0.f ? sv : ALPHA_ * sv;
      float vl = av > 0 ? e : NEG_BIG;
      float w  = __expf(vl - M) * R;
      sFrag[wloc * 512 + (half * 16 + m) * 16 + (jel0 + s)] = f32_to_bf16(w);
    }
    __syncthreads();
    size_t fragIdx = ((size_t)b * WTILES + (jb >> 4) + (t >> 6)) * KTILES + kk;
    async_store_b128(((uint4*)Afrag) + fragIdx * 64 + (t & 63), fbase + t * 16);
    wait_async0();          // LDS tile fully read before next kk refills it
    __syncthreads();
  }
}

// ---------------------------------------------------------------------------
// Kernel 3: per b: C(2048 x 384) = attnT(2048x2048, A) x Wh(2048x384, B)
// bf16 WMMA 16x16x32, f32 accumulate. B panel double-buffered in LDS via
// global_load_async_to_lds_b128 (no VGPR staging -> no spills); A fragments
// streamed from L2 with prefetch. Epilogue: ELU + scatter to out[b,f,n,l].
// Wave owns 1 w-tile x 12 n-tiles (12 v8f accumulators).
// ---------------------------------------------------------------------------
__global__ __launch_bounds__(256)
void k3_gemm(const unsigned short* __restrict__ Afrag,
             const unsigned short* __restrict__ Bfrag,
             float* __restrict__ out) {
  __shared__ uint4 Bsh[2][NTILES * 64];      // 2 x 24 KB

  const int t    = threadIdx.x;
  const int wave = t >> 5, lane = t & 31;
  const int wg   = blockIdx.x, b = blockIdx.y;
  const int wtile = wg * 4 + (wave & 3);
  const int nbase = (wave >> 2) * 12;

  const __bf16* Ab = (const __bf16*)Afrag +
                     (((size_t)b * WTILES + wtile) * KTILES) * 512 + lane * 16;
  const uint4* Bg = (const uint4*)Bfrag + ((size_t)b * KTILES) * NTILES * 64;
  const unsigned bshb = lds_offset(&Bsh[0][0]);

  v8f acc[12];
  #pragma unroll
  for (int q = 0; q < 12; ++q) { v8f z = {}; acc[q] = z; }

  // async-stage chunk 0 into buffer 0
  {
    const uint4* src = Bg;
    #pragma unroll
    for (int k = 0; k < 6; ++k)
      async_load_b128(bshb + (unsigned)(t + k * 256) * 16u, src + t + k * 256);
  }
  wait_async0();
  __syncthreads();

  for (int c = 0; c < KTILES; ++c) {
    const int nb = c & 1;
    if (c + 1 < KTILES) {            // DMA next chunk into the other buffer
      const uint4* src = Bg + (size_t)(c + 1) * NTILES * 64;
      const unsigned dbase = bshb + (unsigned)((nb ^ 1) * NTILES * 64) * 16u;
      #pragma unroll
      for (int k = 0; k < 6; ++k)
        async_load_b128(dbase + (unsigned)(t + k * 256) * 16u, src + t + k * 256);
      __builtin_prefetch(Ab + (size_t)(c + 1) * 512, 0, 3);
    }

    v16bf af = *(const v16bf*)(Ab + (size_t)c * 512);
    const __bf16* bb = (const __bf16*)&Bsh[nb][0];
    // software-pipelined B-fragment reads: issue ds_load q+1 before wmma q
    v16bf bf_cur = *(const v16bf*)(bb + nbase * 512 + lane * 16);
    #pragma unroll
    for (int q = 0; q < 12; ++q) {
      v16bf bf_nxt = (q + 1 < 12)
          ? *(const v16bf*)(bb + (nbase + q + 1) * 512 + lane * 16)
          : bf_cur;
      acc[q] = __builtin_amdgcn_wmma_f32_16x16x32_bf16(
          false, af, false, bf_cur, (short)0, acc[q], false, false);
      bf_cur = bf_nxt;
    }

    wait_async0();                   // next chunk resident in LDS
    __syncthreads();
  }

  // epilogue: C/D layout -> lane L, vgpr r: M = r + (L/16)*8, N = L%16
  const int nl = lane & 15, hlf = lane >> 4;
  #pragma unroll
  for (int q = 0; q < 12; ++q) {
    int nidx = (nbase + q) * 16 + nl;        // packed (l*32 + f)
    int ll = nidx >> 5;
    int ff = nidx & 31;
    float* op = out + (((size_t)b * FO + ff) * NN) * LL + ll;
    #pragma unroll
    for (int r = 0; r < 8; ++r) {
      int w   = wtile * 16 + hlf * 8 + r;
      float x = acc[q][r];
      float y = x > 0.f ? x : (__expf(x) - 1.f);   // ELU
      op[(size_t)w * LL] = y;
    }
  }
}

// ---------------------------------------------------------------------------
extern "C" void kernel_launch(void* const* d_in, const int* in_sizes, int n_in,
                              void* d_out, int out_size, void* d_ws, size_t ws_size,
                              hipStream_t stream) {
  (void)in_sizes; (void)n_in; (void)out_size; (void)ws_size;
  const float* h   = (const float*)d_in[0];
  const int*   adj = (const int*)d_in[1];
  const float* W   = (const float*)d_in[2];
  const float* a   = (const float*)d_in[3];
  float* out = (float*)d_out;

  char* ws = (char*)d_ws;
  const size_t szB = (size_t)B_ * KTILES * NTILES * 1024;   // 12 MB  Wh frags
  const size_t szA = (size_t)B_ * WTILES * KTILES * 1024;   // 64 MB  attnT frags
  unsigned short* Bfrag = (unsigned short*)ws;
  unsigned short* Afrag = (unsigned short*)(ws + szB);
  float* s1 = (float*)(ws + szB + szA);
  float* s2 = s1 + (size_t)B_ * NN;

  k1_wh  <<<dim3(KTILES, B_),   384, 0, stream>>>(h, W, a, Bfrag, s1, s2);
  k2_attn<<<dim3(NN / 64, B_),  256, 0, stream>>>(adj, s1, s2, Afrag);
  k3_gemm<<<dim3(WTILES / 4, B_), 256, 0, stream>>>(Afrag, Bfrag, out);
}
